// VanishingNet_41875931136703
// MI455X (gfx1250) — matrix operand order repaired and checked by reading, not compiled, not run on hardware
//
#include <hip/hip_runtime.h>
#include <hip/hip_bf16.h>
#include <math.h>

#define BB 2
#define CC 128
#define NPIX 16384
#define HTH 184
#define HTW 180
#define NBINS 33120
#define KHT 8
#define KSPH 16
#define EPSV 1e-5f
#define PRED_STRIDE 1984

typedef __attribute__((ext_vector_type(16))) _Float16 v16h;
typedef __attribute__((ext_vector_type(8)))  float    v8f;
typedef __attribute__((ext_vector_type(4)))  int      v4i;

union Frag16 { v16h v; uint4 q[2]; };

__device__ __forceinline__ v8f wmma_f16(const Frag16& a, const Frag16& b, v8f c) {
  return __builtin_amdgcn_wmma_f32_16x16x32_f16(false, a.v, false, b.v, (short)0, c, false, false);
}

// ----- async global->LDS staging (CDNA5 GLOBAL_LOAD_ASYNC_TO_LDS_B128), guarded -----
#if defined(__has_builtin)
#  if __has_builtin(__builtin_amdgcn_global_load_async_to_lds_b128) && \
      __has_builtin(__builtin_amdgcn_s_wait_asynccnt)
#    define HAVE_ASYNC_LDS 1
#  endif
#endif
#ifndef HAVE_ASYNC_LDS
#  define HAVE_ASYNC_LDS 0
#endif

#if HAVE_ASYNC_LDS
typedef __attribute__((address_space(1))) v4i* g_v4i_ptr;
typedef __attribute__((address_space(3))) v4i* l_v4i_ptr;
#endif

__device__ __forceinline__ void lds_copy16(void* l, const void* g) {
#if HAVE_ASYNC_LDS
  __builtin_amdgcn_global_load_async_to_lds_b128(
      (g_v4i_ptr)(void*)g, (l_v4i_ptr)l, 0, 0);
#else
  *(uint4*)l = *(const uint4*)g;
#endif
}

__device__ __forceinline__ void lds_copy_fence() {
#if HAVE_ASYNC_LDS
  __builtin_amdgcn_s_wait_asynccnt(0);
#endif
}

// ---------------- BN2d stats: input channel-major [b][c][plane] ----------------
__global__ void bn2d_stats_kernel(const float* __restrict__ x, float* __restrict__ mean,
                                  float* __restrict__ rstd, int plane) {
  int c = blockIdx.x;
  float s = 0.f, s2 = 0.f;
  for (int b = 0; b < BB; ++b) {
    const float* p = x + ((size_t)b * CC + c) * (size_t)plane;
    for (int i = threadIdx.x; i < plane; i += 256) { float v = p[i]; s += v; s2 += v * v; }
  }
  __shared__ float rs[256], rq[256];
  rs[threadIdx.x] = s; rq[threadIdx.x] = s2;
  __syncthreads();
  for (int o = 128; o > 0; o >>= 1) {
    if (threadIdx.x < o) { rs[threadIdx.x] += rs[threadIdx.x + o]; rq[threadIdx.x] += rq[threadIdx.x + o]; }
    __syncthreads();
  }
  if (threadIdx.x == 0) {
    float n = (float)plane * BB;
    float m = rs[0] / n;
    float var = rq[0] / n - m * m;
    mean[c] = m; rstd[c] = rsqrtf(var + EPSV);
  }
}

// ---------------- BN2d stats: input channel-last [b][plane][c] ----------------
__global__ void bn2d_stats_cl_kernel(const float* __restrict__ x, float* __restrict__ mean,
                                     float* __restrict__ rstd, int plane) {
  int c = blockIdx.x;
  float s = 0.f, s2 = 0.f;
  for (int b = 0; b < BB; ++b) {
    const float* p = x + (size_t)b * plane * CC + c;
    for (int i = threadIdx.x; i < plane; i += 256) { float v = p[(size_t)i * CC]; s += v; s2 += v * v; }
  }
  __shared__ float rs[256], rq[256];
  rs[threadIdx.x] = s; rq[threadIdx.x] = s2;
  __syncthreads();
  for (int o = 128; o > 0; o >>= 1) {
    if (threadIdx.x < o) { rs[threadIdx.x] += rs[threadIdx.x + o]; rq[threadIdx.x] += rq[threadIdx.x + o]; }
    __syncthreads();
  }
  if (threadIdx.x == 0) {
    float n = (float)plane * BB;
    float m = rs[0] / n;
    float var = rq[0] / n - m * m;
    mean[c] = m; rstd[c] = rsqrtf(var + EPSV);
  }
}

// image: in [b][c][p] -> BN+ReLU -> out channel-last [b][p][c]
__global__ void bn2d_apply_relu_tr_kernel(const float* __restrict__ x, float* __restrict__ y,
                                          const float* __restrict__ mean, const float* __restrict__ rstd,
                                          const float* __restrict__ g, const float* __restrict__ be,
                                          long long total) {
  long long i = (long long)blockIdx.x * 256 + threadIdx.x;   // output index, c fastest
  if (i >= total) return;
  int c = (int)(i & 127);
  long long r = i >> 7;
  int p = (int)(r % NPIX);
  int b = (int)(r / NPIX);
  float v = x[((size_t)b * CC + c) * NPIX + p];
  v = g[c] * (v - mean[c]) * rstd[c] + be[c];
  y[i] = v > 0.f ? v : 0.f;
}

// channel-last in-place BN+ReLU: [b][plane][c]
__global__ void bn2d_apply_relu_cl_kernel(const float* __restrict__ x, float* __restrict__ y,
                                          const float* __restrict__ mean, const float* __restrict__ rstd,
                                          const float* __restrict__ g, const float* __restrict__ be,
                                          long long total) {
  long long i = (long long)blockIdx.x * 256 + threadIdx.x;
  if (i >= total) return;
  int c = (int)(i & 127);
  float v = g[c] * (x[i] - mean[c]) * rstd[c] + be[c];
  y[i] = v > 0.f ? v : 0.f;
}

// ---------------- Hough gather: x_relu [b][pix][c] -> htA f16 [b][bin][c] ----------------
__global__ void ht_gather_kernel(const float* __restrict__ xr, const int* __restrict__ idx,
                                 const float* __restrict__ w, _Float16* __restrict__ outp,
                                 long long total) {
  long long t = (long long)blockIdx.x * 256 + threadIdx.x;
  if (t >= total) return;
  int c = (int)(t & 127);
  long long r = t >> 7;
  int bin = (int)(r % NBINS);
  int b = (int)(r / NBINS);
  const float* xp = xr + (size_t)b * NPIX * CC + c;
  const int* ip = idx + (size_t)bin * KHT;
  const float* wp = w + (size_t)bin * KHT;
  float acc = 0.f;
#pragma unroll
  for (int k = 0; k < KHT; ++k) acc += xp[(size_t)ip[k] * CC] * wp[k];
  outp[t] = (_Float16)acc;
}

// ---------------- weight conversions ----------------
__global__ void conv_w_transpose_kernel(const float* __restrict__ w, _Float16* __restrict__ wt) {
  // w [3][3][ic][oc] -> wt [t][oc][ic]  (f16)
  int i = blockIdx.x * 256 + threadIdx.x;
  if (i >= 9 * CC * CC) return;
  int ic = i & 127;
  int oc = (i >> 7) & 127;
  int t = i >> 14;
  wt[i] = (_Float16)w[((size_t)t * CC + ic) * CC + oc];
}

__global__ void f32_to_f16_kernel(const float* __restrict__ in, _Float16* __restrict__ outp, int n) {
  int i = blockIdx.x * 256 + threadIdx.x;
  if (i < n) outp[i] = (_Float16)in[i];
}

__global__ void dgcn_w_transpose_kernel(const float* __restrict__ w, _Float16* __restrict__ wt) {
  // w [12][128][64] -> wt [12][64][128]
  int i = blockIdx.x * 256 + threadIdx.x;
  if (i >= 12 * 64 * 128) return;
  int k = i & 127;
  int o = (i >> 7) & 63;
  int si = i >> 13;
  wt[i] = (_Float16)w[((size_t)si * 128 + k) * 64 + o];
}

// ---------------- 3x3 conv as implicit-GEMM WMMA; out hc channel-last [b][bin][oc] ----------------
__global__ __launch_bounds__(256)
void htconv_wmma_kernel(const _Float16* __restrict__ htA, const _Float16* __restrict__ wt,
                        float* __restrict__ hc) {
  __shared__ __align__(16) _Float16 sA[64 * 128];
  __shared__ __align__(16) _Float16 sW[128 * 128];
  int b = blockIdx.y;
  int pixBase = blockIdx.x * 64;
  int tid = threadIdx.x;
  int wid = tid >> 5, lane = tid & 31, l = lane & 15, hi = lane >> 4;
  int mG = wid & 3, nG = wid >> 2;   // 4 M-groups x 2 N-groups (of 64)
  v8f acc[4];
#pragma unroll
  for (int i = 0; i < 4; ++i)
#pragma unroll
    for (int j = 0; j < 8; ++j) acc[i][j] = 0.f;

  for (int t = 0; t < 9; ++t) {
    int dy = t / 3 - 1, dx = t % 3 - 1;
    __syncthreads();
    {
      int r = tid >> 2, ch = (tid & 3) * 32;
      int p = pixBase + r;
      bool valid = false;
      const _Float16* src = htA;
      if (p < NBINS) {
        int py = p / HTW, px = p - py * HTW;
        int sy = py + dy, sx = px + dx;
        if (sy >= 0 && sy < HTH && sx >= 0 && sx < HTW) {
          valid = true;
          src = htA + ((size_t)b * NBINS + (size_t)sy * HTW + sx) * CC + ch;
        }
      }
      _Float16* dst = sA + r * 128 + ch;
      if (valid) {
#pragma unroll
        for (int j = 0; j < 4; ++j) lds_copy16(dst + j * 8, src + j * 8);
      } else {
        uint4 z = make_uint4(0u, 0u, 0u, 0u);
#pragma unroll
        for (int j = 0; j < 4; ++j) ((uint4*)dst)[j] = z;
      }
      const _Float16* wsrc = wt + (size_t)t * CC * CC;
#pragma unroll
      for (int i = 0; i < 8; ++i)
        lds_copy16(sW + (size_t)(tid + i * 256) * 8, wsrc + (size_t)(tid + i * 256) * 8);
    }
    if (tid == 0 && t < 8) __builtin_prefetch(wt + (size_t)(t + 1) * CC * CC, 0, 1);
    lds_copy_fence();
    __syncthreads();
#pragma unroll
    for (int kb = 0; kb < 4; ++kb) {
      int k0 = kb * 32;
      Frag16 a;
      const _Float16* ap = sA + (mG * 16 + l) * 128;
      a.q[0] = *(const uint4*)(ap + k0 + hi * 8);
      a.q[1] = *(const uint4*)(ap + k0 + 16 + hi * 8);
#pragma unroll
      for (int nb = 0; nb < 4; ++nb) {
        Frag16 bf;
        const uint4* bp = (const uint4*)(sW + (nG * 64 + nb * 16 + l) * 128 + k0 + hi * 8);
        bf.q[0] = bp[0]; bf.q[1] = bp[2];
        acc[nb] = wmma_f16(a, bf, acc[nb]);
      }
    }
  }
  // store channel-last: hc[(b*NBINS+pix)*128 + oc]  (lanes contiguous in oc)
#pragma unroll
  for (int nb = 0; nb < 4; ++nb) {
    int oc = nG * 64 + nb * 16 + l;
#pragma unroll
    for (int r = 0; r < 8; ++r) {
      int pix = pixBase + mG * 16 + hi * 8 + r;
      if (pix < NBINS) hc[((size_t)b * NBINS + pix) * CC + oc] = acc[nb][r];
    }
  }
}

// ---------------- sphere gather: x_htf channel-last -> f16 [b][j][c] ----------------
__global__ void sphere_gather_kernel(const float* __restrict__ xhtf, const int* __restrict__ sidx,
                                     const float* __restrict__ sw, const int* __restrict__ ind,
                                     _Float16* __restrict__ outp, int S, long long total) {
  long long t = (long long)blockIdx.x * 256 + threadIdx.x;
  if (t >= total) return;
  int c = (int)(t & 127);
  long long r = t >> 7;
  int j = (int)(r % S);
  int b = (int)(r / S);
  int node = ind[(size_t)b * S + j];
  const float* xp = xhtf + (size_t)b * NBINS * CC + c;
  const int* ip = sidx + (size_t)node * KSPH;
  const float* wp = sw + (size_t)node * KSPH;
  float acc = 0.f;
#pragma unroll
  for (int k = 0; k < KSPH; ++k) acc += xp[(size_t)ip[k] * CC] * wp[k];
  outp[t] = (_Float16)acc;
}

// ---------------- tiled WMMA GEMM: out[M][64] = A[M][128] * WT[64][128]^T + bias ----------------
__global__ __launch_bounds__(256)
void gemm_f16_k128_n64_kernel(const _Float16* __restrict__ A, const _Float16* __restrict__ WT,
                              const float* __restrict__ bias, float* __restrict__ outp,
                              int M, int doRelu) {
  __shared__ __align__(16) _Float16 sA[128 * 128];
  __shared__ __align__(16) _Float16 sW[64 * 128];
  int tid = threadIdx.x;
  int wid = tid >> 5, lane = tid & 31, l = lane & 15, hi = lane >> 4;
  int mBase = blockIdx.x * 128;
#pragma unroll
  for (int i = 0; i < 8; ++i) {
    int u = tid + i * 256;
    int r = u >> 4, cpart = u & 15;
    if (mBase + r < M) {
      lds_copy16(sA + (size_t)u * 8, A + (size_t)(mBase + r) * 128 + cpart * 8);
    } else {
      ((uint4*)sA)[u] = make_uint4(0u, 0u, 0u, 0u);
    }
  }
#pragma unroll
  for (int i = 0; i < 4; ++i)
    lds_copy16(sW + (size_t)(tid + i * 256) * 8, WT + (size_t)(tid + i * 256) * 8);
  lds_copy_fence();
  __syncthreads();
  v8f acc[4];
#pragma unroll
  for (int i = 0; i < 4; ++i)
#pragma unroll
    for (int j = 0; j < 8; ++j) acc[i][j] = 0.f;
#pragma unroll
  for (int kb = 0; kb < 4; ++kb) {
    int k0 = kb * 32;
    Frag16 a;
    const _Float16* ap = sA + (wid * 16 + l) * 128;
    a.q[0] = *(const uint4*)(ap + k0 + hi * 8);
    a.q[1] = *(const uint4*)(ap + k0 + 16 + hi * 8);
#pragma unroll
    for (int nb = 0; nb < 4; ++nb) {
      Frag16 bf;
      const uint4* bp = (const uint4*)(sW + (nb * 16 + l) * 128 + k0 + hi * 8);
      bf.q[0] = bp[0]; bf.q[1] = bp[2];
      acc[nb] = wmma_f16(a, bf, acc[nb]);
    }
  }
#pragma unroll
  for (int nb = 0; nb < 4; ++nb) {
    int n = nb * 16 + l;
    float bv = bias[n];
#pragma unroll
    for (int r = 0; r < 8; ++r) {
      int m = mBase + wid * 16 + hi * 8 + r;
      if (m < M) {
        float v = acc[nb][r] + bv;
        if (doRelu) v = v > 0.f ? v : 0.f;
        outp[(size_t)m * 64 + n] = v;
      }
    }
  }
}

// ---------------- BN1d over M rows per feature (64) ----------------
__global__ void bn1d_stats_kernel(const float* __restrict__ y, float* __restrict__ mean,
                                  float* __restrict__ rstd, int M) {
  int n = blockIdx.x;
  float s = 0.f, s2 = 0.f;
  for (int i = threadIdx.x; i < M; i += 256) { float v = y[(size_t)i * 64 + n]; s += v; s2 += v * v; }
  __shared__ float rs[256], rq[256];
  rs[threadIdx.x] = s; rq[threadIdx.x] = s2;
  __syncthreads();
  for (int o = 128; o > 0; o >>= 1) {
    if (threadIdx.x < o) { rs[threadIdx.x] += rs[threadIdx.x + o]; rq[threadIdx.x] += rq[threadIdx.x + o]; }
    __syncthreads();
  }
  if (threadIdx.x == 0) {
    float nn = (float)M;
    float m = rs[0] / nn;
    float var = rq[0] / nn - m * m;
    mean[n] = m; rstd[n] = rsqrtf(var + EPSV);
  }
}

__global__ void bn1d_apply_relu_kernel(const float* __restrict__ y, float* __restrict__ x,
                                       const float* __restrict__ mean, const float* __restrict__ rstd,
                                       const float* __restrict__ g, const float* __restrict__ be,
                                       long long total) {
  long long i = (long long)blockIdx.x * 256 + threadIdx.x;
  if (i >= total) return;
  int n = (int)(i & 63);
  float v = g[n] * (y[i] - mean[n]) * rstd[n] + be[n];
  x[i] = v > 0.f ? v : 0.f;
}

// ---------------- EdgeConv helpers ----------------
__global__ void edge_feat_kernel(const float* __restrict__ x, const int* __restrict__ edge,
                                 _Float16* __restrict__ A, int nNode, int nEdge, long long total) {
  long long t = (long long)blockIdx.x * 256 + threadIdx.x;
  if (t >= total) return;
  int c = (int)(t & 127);
  long long r = t >> 7;
  int e = (int)(r % nEdge);
  int g = (int)(r / nEdge);
  const int* eg = edge + (size_t)g * 2 * nEdge;
  int ctr = eg[e];
  int nbr = eg[nEdge + e];
  const float* xg = x + (size_t)g * nNode * 64;
  float v;
  if (c < 64) v = xg[(size_t)ctr * 64 + c];
  else { int cc2 = c - 64; v = xg[(size_t)nbr * 64 + cc2] - xg[(size_t)ctr * 64 + cc2]; }
  A[t] = (_Float16)v;
}

__global__ void edge_max_kernel(const float* __restrict__ h, float* __restrict__ x,
                                long long total) {
  long long t = (long long)blockIdx.x * 256 + threadIdx.x;
  if (t >= total) return;
  long long rn = t >> 6;
  int o = (int)(t & 63);
  const float* hp = h + (size_t)rn * 8 * 64 + o;
  float m = hp[0];
#pragma unroll
  for (int i = 1; i < 8; ++i) m = fmaxf(m, hp[(size_t)i * 64]);
  x[t] = m;
}

__global__ void head_kernel(const float* __restrict__ x, const float* __restrict__ hw,
                            const float* __restrict__ hb, float* __restrict__ logits,
                            float* __restrict__ outp, int nNode, int branches, int predBase,
                            long long total) {
  long long t = (long long)blockIdx.x * 256 + threadIdx.x;
  if (t >= total) return;
  int node = (int)(t % nNode);
  int g = (int)(t / nNode);
  const float* xp = x + (size_t)t * 64;
  float acc = hb[0];
#pragma unroll
  for (int o = 0; o < 64; ++o) acc += xp[o] * hw[o];
  logits[t] = acc;
  int b = g / branches, rr = g % branches;
  outp[6 + (size_t)b * PRED_STRIDE + predBase + rr * nNode + node] = 1.f / (1.f + expf(-acc));
}

__global__ void bce_kernel(const float* __restrict__ logits, const int* __restrict__ tgt,
                           float* __restrict__ acc4, int n) {
  float sp = 0.f, sn = 0.f, cp = 0.f, cn = 0.f;
  for (int i = blockIdx.x * 256 + threadIdx.x; i < n; i += gridDim.x * 256) {
    float z = logits[i];
    float tt = (float)tgt[i];
    float l = fmaxf(z, 0.f) - z * tt + log1pf(expf(-fabsf(z)));
    if (tt > 0.5f) { sp += l; cp += 1.f; } else { sn += l; cn += 1.f; }
  }
  __shared__ float r0[256], r1[256], r2[256], r3[256];
  int tid = threadIdx.x;
  r0[tid] = sp; r1[tid] = sn; r2[tid] = cp; r3[tid] = cn;
  __syncthreads();
  for (int o = 128; o > 0; o >>= 1) {
    if (tid < o) { r0[tid] += r0[tid + o]; r1[tid] += r1[tid + o]; r2[tid] += r2[tid + o]; r3[tid] += r3[tid + o]; }
    __syncthreads();
  }
  if (tid == 0) {
    atomicAdd(acc4 + 0, r0[0]); atomicAdd(acc4 + 1, r1[0]);
    atomicAdd(acc4 + 2, r2[0]); atomicAdd(acc4 + 3, r3[0]);
  }
}

__global__ void zero_kernel(float* __restrict__ p, int n) {
  int i = blockIdx.x * 256 + threadIdx.x;
  if (i < n) p[i] = 0.f;
}

__global__ void finalize_kernel(const float* __restrict__ acc, float* __restrict__ outp) {
  int i = threadIdx.x;
  if (i < 6) {
    const float* a = acc + (i >> 1) * 4;
    float num = (i & 1) ? a[1] : a[0];
    float den = fmaxf((i & 1) ? a[3] : a[2], 1.f);
    outp[i] = num / den;
  }
}

static inline unsigned cdivu(long long a, long long b) { return (unsigned)((a + b - 1) / b); }

extern "C" void kernel_launch(void* const* d_in, const int* in_sizes, int n_in,
                              void* d_out, int out_size, void* d_ws, size_t ws_size,
                              hipStream_t stream) {
  const float* image     = (const float*)d_in[0];
  const float* bn_gamma  = (const float*)d_in[1];
  const float* bn_beta   = (const float*)d_in[2];
  const int*   ht_idx    = (const int*)d_in[3];
  const float* ht_w      = (const float*)d_in[4];
  const float* w_htconv  = (const float*)d_in[5];
  const float* htbn_gamma= (const float*)d_in[6];
  const float* htbn_beta = (const float*)d_in[7];
  const int*   sph_idx   = (const int*)d_in[8];
  const float* sph_w     = (const float*)d_in[9];
  const float* w_sc      = (const float*)d_in[10];
  const float* b_sc      = (const float*)d_in[11];
  const float* scbn_gamma= (const float*)d_in[12];
  const float* scbn_beta = (const float*)d_in[13];
  const float* dgcn_w    = (const float*)d_in[14];
  const float* dgcn_b    = (const float*)d_in[15];
  const float* dgcn_hw   = (const float*)d_in[16];
  const float* dgcn_hb   = (const float*)d_in[17];
  const int*   ind0      = (const int*)d_in[18];
  const int*   ind1      = (const int*)d_in[19];
  const int*   ind2      = (const int*)d_in[20];
  const int*   edge0     = (const int*)d_in[21];
  const int*   edge1     = (const int*)d_in[22];
  const int*   edge2     = (const int*)d_in[23];
  const int*   target0   = (const int*)d_in[24];
  const int*   target1   = (const int*)d_in[25];
  const int*   target2   = (const int*)d_in[26];
  float* out = (float*)d_out;

  char* base = (char*)d_ws;
  size_t off = 0;
  auto take = [&](size_t bytes) -> char* {
    char* p = base + off;
    off = (off + bytes + 255) & ~(size_t)255;
    return p;
  };
  float*    meanA   = (float*)take(CC * 4);
  float*    rstdA   = (float*)take(CC * 4);
  float*    x_relu  = (float*)take((size_t)BB * CC * NPIX * 4);   // channel-last [b][pix][c]
  _Float16* htA     = (_Float16*)take((size_t)BB * NBINS * CC * 2);
  _Float16* wtConv  = (_Float16*)take((size_t)9 * CC * CC * 2);
  float*    hc      = (float*)take((size_t)BB * CC * NBINS * 4);  // channel-last [b][bin][c]
  float*    meanB   = (float*)take(CC * 4);
  float*    rstdB   = (float*)take(CC * 4);
  _Float16* wscH    = (_Float16*)take((size_t)3 * 64 * CC * 2);
  _Float16* dgcnWT  = (_Float16*)take((size_t)12 * 64 * CC * 2);
  _Float16* sphA    = (_Float16*)take((size_t)BB * 1024 * CC * 2);
  float*    ysc     = (float*)take((size_t)2048 * 64 * 4);
  float*    bnM     = (float*)take(64 * 4);
  float*    bnR     = (float*)take(64 * 4);
  float*    xn0     = (float*)take((size_t)2048 * 64 * 4);
  float*    xn1     = (float*)take((size_t)2048 * 64 * 4);
  _Float16* edgeA   = (_Float16*)take((size_t)16384 * CC * 2);
  float*    hbuf    = (float*)take((size_t)16384 * 64 * 4);
  float*    logitsB = (float*)take(2048 * 4);
  float*    accBuf  = (float*)take(12 * 4);

  dim3 blk(256);

  // image BN + ReLU (stats on channel-major input, output channel-last)
  bn2d_stats_kernel<<<CC, blk, 0, stream>>>(image, meanA, rstdA, NPIX);
  long long tImg = (long long)BB * CC * NPIX;
  bn2d_apply_relu_tr_kernel<<<cdivu(tImg, 256), blk, 0, stream>>>(image, x_relu, meanA, rstdA,
                                                                  bn_gamma, bn_beta, tImg);
  // Hough gather (f16, channel-last, coalesced gathers)
  long long tHt = (long long)BB * NBINS * CC;
  ht_gather_kernel<<<cdivu(tHt, 256), blk, 0, stream>>>(x_relu, ht_idx, ht_w, htA, tHt);
  // conv weights -> f16 [t][oc][ic]
  conv_w_transpose_kernel<<<cdivu(9 * CC * CC, 256), blk, 0, stream>>>(w_htconv, wtConv);
  // 3x3 conv via WMMA implicit GEMM (async LDS staging)
  dim3 cg((NBINS + 63) / 64, BB);
  htconv_wmma_kernel<<<cg, blk, 0, stream>>>(htA, wtConv, hc);
  // BN + ReLU on conv output (channel-last, in place) -> x_htf
  bn2d_stats_cl_kernel<<<CC, blk, 0, stream>>>(hc, meanB, rstdB, NBINS);
  long long tHc = (long long)BB * CC * NBINS;
  bn2d_apply_relu_cl_kernel<<<cdivu(tHc, 256), blk, 0, stream>>>(hc, hc, meanB, rstdB,
                                                                 htbn_gamma, htbn_beta, tHc);
  // small weight conversions
  f32_to_f16_kernel<<<cdivu(3 * 64 * CC, 256), blk, 0, stream>>>(w_sc, wscH, 3 * 64 * CC);
  dgcn_w_transpose_kernel<<<cdivu(12 * 64 * CC, 256), blk, 0, stream>>>(dgcn_w, dgcnWT);

  zero_kernel<<<1, 32, 0, stream>>>(accBuf, 12);

  const int S_[3]        = {1024, 768, 192};
  const int nNode_[3]    = {1024, 256, 64};
  const int branches_[3] = {1, 3, 3};
  const int predBase_[3] = {0, 1024, 1792};
  const int* indPtr[3]  = {ind0, ind1, ind2};
  const int* edgePtr[3] = {edge0, edge1, edge2};
  const int* tgtPtr[3]  = {target0, target1, target2};

  for (int s = 0; s < 3; ++s) {
    int S = S_[s], nNode = nNode_[s];
    int Bg = BB * branches_[s];
    int nEdge = nNode * 8;
    int M = BB * S;   // == Bg*nNode
    long long tg = (long long)M * CC;
    sphere_gather_kernel<<<cdivu(tg, 256), blk, 0, stream>>>(hc, sph_idx, sph_w, indPtr[s], sphA, S, tg);
    gemm_f16_k128_n64_kernel<<<(M + 127) / 128, blk, 0, stream>>>(
        sphA, wscH + (size_t)s * 64 * CC, b_sc + s * 64, ysc, M, 0);
    bn1d_stats_kernel<<<64, blk, 0, stream>>>(ysc, bnM, bnR, M);
    long long tb = (long long)M * 64;
    bn1d_apply_relu_kernel<<<cdivu(tb, 256), blk, 0, stream>>>(ysc, xn0, bnM, bnR,
                                                               scbn_gamma + s * 64, scbn_beta + s * 64, tb);
    float* cur = xn0;
    float* nxt = xn1;
    int Me = Bg * nEdge;
    for (int layer = 0; layer < 4; ++layer) {
      long long te = (long long)Me * CC;
      edge_feat_kernel<<<cdivu(te, 256), blk, 0, stream>>>(cur, edgePtr[s], edgeA, nNode, nEdge, te);
      gemm_f16_k128_n64_kernel<<<(Me + 127) / 128, blk, 0, stream>>>(
          edgeA, dgcnWT + (size_t)(s * 4 + layer) * 64 * CC, dgcn_b + (s * 4 + layer) * 64, hbuf, Me, 1);
      long long tm = (long long)Bg * nNode * 64;
      edge_max_kernel<<<cdivu(tm, 256), blk, 0, stream>>>(hbuf, nxt, tm);
      float* tswap = cur; cur = nxt; nxt = tswap;
    }
    long long th = (long long)Bg * nNode;
    head_kernel<<<cdivu(th, 256), blk, 0, stream>>>(cur, dgcn_hw + s * 64, dgcn_hb + s,
                                                    logitsB, out, nNode, branches_[s], predBase_[s], th);
    bce_kernel<<<8, blk, 0, stream>>>(logitsB, tgtPtr[s], accBuf + s * 4, (int)th);
  }
  finalize_kernel<<<1, 32, 0, stream>>>(accBuf, out);

  (void)in_sizes; (void)n_in; (void)out_size; (void)ws_size;
}
